// MaskedLinearMultinotes_37194416783525
// MI455X (gfx1250) — matrix operand order repaired and checked by reading, not compiled, not run on hardware
//
#include <hip/hip_runtime.h>

// CDNA5 / gfx1250, wave32. bf16 WMMA types per CDNA5_HIP.md.
typedef __attribute__((ext_vector_type(16))) __bf16 v16bf;
typedef __attribute__((ext_vector_type(8)))  __bf16 v8bf;
typedef __attribute__((ext_vector_type(8)))  float  v8f;

constexpr int B_ = 32, G_ = 16, O_ = 256, N_ = 88, T_ = 256;
constexpr int WAVES = 8;               // 8 waves / workgroup, one 'o' per wave
constexpr int OB    = O_ / WAVES;      // 32 o-chunks per n

union AFrag { v16bf v; v8bf h[2]; };   // A fragment = two contiguous 8-elem K runs

__global__ __launch_bounds__(256)
void masked_multinote_wmma(const float* __restrict__ x,
                           const float* __restrict__ y,
                           const float* __restrict__ w,
                           const float* __restrict__ mask,
                           const float* __restrict__ bias,
                           float* __restrict__ out)
{
    __shared__ __bf16 xbf[B_ * T_];    // x[:, n, :] as bf16, 16 KB
    __shared__ float  ylds[B_ * G_];   // y, 2 KB (512 entries)

    const int tid  = threadIdx.x;
    const int wave = tid >> 5;
    const int lane = tid & 31;
    const int h    = lane >> 4;        // K-half selector (ISA A/B lane split)
    const int g    = lane & 15;        // B-matrix N index == genre; also A-matrix M index

    const int n = blockIdx.x / OB;
    const int o = (blockIdx.x % OB) * WAVES + wave;

    // ---- stage x[:, n, :] -> LDS bf16 (coalesced float4 reads; x lives in L2) ----
    for (int i = tid; i < (B_ * T_) / 4; i += 256) {
        const int b = i >> 6;                  // 64 float4 per (b) row
        const int t = (i & 63) << 2;
        const float4 v = *(const float4*)(x + ((size_t)b * N_ + n) * T_ + t);
        xbf[b * T_ + t + 0] = (__bf16)v.x;
        xbf[b * T_ + t + 1] = (__bf16)v.y;
        xbf[b * T_ + t + 2] = (__bf16)v.z;
        xbf[b * T_ + t + 3] = (__bf16)v.w;
    }
    // y has B*G = 512 entries; 256 threads stage two each.
    ylds[tid]       = y[tid];
    ylds[tid + 256] = y[tid + 256];
    __syncthreads();

    // ---- z[b,g] = sum_t xbf[b,t] * (w[g,o,n,t] * mask[o,t]), via bf16 WMMA ----
    const size_t gstride = (size_t)O_ * N_ * T_;          // weight g-stride
    const float* wbase = w + ((size_t)o * N_ + n) * T_;   // g = 0 row for this (o,n)
    const float* mrow  = mask + (size_t)o * T_;

    v8f acc0 = {};   // b = 0..15
    v8f acc1 = {};   // b = 16..31

#pragma unroll
    for (int ks = 0; ks < T_ / 32; ++ks) {
        const int t0 = ks * 32;

        // B fragment (32x16, K=t, N=g): lane (g,h) holds 16 contiguous t values
        // at t0+16h .. t0+16h+15  -> four perfectly aligned float4 loads.
        const float* wp = wbase + (size_t)g * gstride + t0 + 16 * h;
        const float* mp = mrow + t0 + 16 * h;
        const float4 w0 = *(const float4*)(wp + 0);
        const float4 w1 = *(const float4*)(wp + 4);
        const float4 w2 = *(const float4*)(wp + 8);
        const float4 w3 = *(const float4*)(wp + 12);
        const float4 m0 = *(const float4*)(mp + 0);
        const float4 m1 = *(const float4*)(mp + 4);
        const float4 m2 = *(const float4*)(mp + 8);
        const float4 m3 = *(const float4*)(mp + 12);

        // Prefetch the weight stream 2 iterations (256 B) ahead, WGP scope
        // (locality 3 -> pull into all cache levels; SYS scope skips the WGP$).
        if (ks + 2 < T_ / 32) __builtin_prefetch(wp + 64, 0, 3);

        v16bf bf;
        bf[0]  = (__bf16)(w0.x * m0.x);  bf[1]  = (__bf16)(w0.y * m0.y);
        bf[2]  = (__bf16)(w0.z * m0.z);  bf[3]  = (__bf16)(w0.w * m0.w);
        bf[4]  = (__bf16)(w1.x * m1.x);  bf[5]  = (__bf16)(w1.y * m1.y);
        bf[6]  = (__bf16)(w1.z * m1.z);  bf[7]  = (__bf16)(w1.w * m1.w);
        bf[8]  = (__bf16)(w2.x * m2.x);  bf[9]  = (__bf16)(w2.y * m2.y);
        bf[10] = (__bf16)(w2.z * m2.z);  bf[11] = (__bf16)(w2.w * m2.w);
        bf[12] = (__bf16)(w3.x * m3.x);  bf[13] = (__bf16)(w3.y * m3.y);
        bf[14] = (__bf16)(w3.z * m3.z);  bf[15] = (__bf16)(w3.w * m3.w);

        // A fragments (16x32, M=b, K=t) from LDS: lane m=g holds
        // K runs {t0+16h..+7} and {t0+16h+8..+7} -> two 16B ds loads per tile.
        AFrag a0, a1;
        a0.h[0] = *(const v8bf*)(xbf + g * T_ + t0 + 16 * h);
        a0.h[1] = *(const v8bf*)(xbf + g * T_ + t0 + 16 * h + 8);
        a1.h[0] = *(const v8bf*)(xbf + (g + 16) * T_ + t0 + 16 * h);
        a1.h[1] = *(const v8bf*)(xbf + (g + 16) * T_ + t0 + 16 * h + 8);

        acc0 = __builtin_amdgcn_wmma_f32_16x16x32_bf16(
                   false, a0.v, false, bf, (short)0, acc0, false, false);
        acc1 = __builtin_amdgcn_wmma_f32_16x16x32_bf16(
                   false, a1.v, false, bf, (short)0, acc1, false, false);
    }

    // ---- out[b,n,o] = sum_g y[b,g] * z[b,g] + bias[o] ----
    // C/D layout: VGPR r, lane L -> M = r + 8*(L>>4), N = L&15 (= g).
    const float bo = bias[o];
#pragma unroll
    for (int r = 0; r < 8; ++r) {
        {
            const int b = 8 * h + r;
            float p = acc0[r] * ylds[b * G_ + g];
            p += __shfl_xor(p, 1);
            p += __shfl_xor(p, 2);
            p += __shfl_xor(p, 4);
            p += __shfl_xor(p, 8);
            if (g == 0) out[((size_t)b * N_ + n) * O_ + o] = p + bo;
        }
        {
            const int b = 16 + 8 * h + r;
            float q = acc1[r] * ylds[b * G_ + g];
            q += __shfl_xor(q, 1);
            q += __shfl_xor(q, 2);
            q += __shfl_xor(q, 4);
            q += __shfl_xor(q, 8);
            if (g == 0) out[((size_t)b * N_ + n) * O_ + o] = q + bo;
        }
    }
}

extern "C" void kernel_launch(void* const* d_in, const int* in_sizes, int n_in,
                              void* d_out, int out_size, void* d_ws, size_t ws_size,
                              hipStream_t stream) {
    const float* x    = (const float*)d_in[0];  // [B, N, T]
    const float* y    = (const float*)d_in[1];  // [B, G]
    const float* w    = (const float*)d_in[2];  // [G, O, N, T]
    const float* mask = (const float*)d_in[3];  // [O, T]
    const float* bias = (const float*)d_in[4];  // [O]
    float* out = (float*)d_out;                 // [B, N, O]

    dim3 grid(N_ * OB);   // 88 * 32 = 2816 workgroups
    dim3 block(256);      // 8 waves; one 'o' per wave
    masked_multinote_wmma<<<grid, block, 0, stream>>>(x, y, w, mask, bias, out);
}
